// GraphNetwork_64424509440356
// MI455X (gfx1250) — matrix operand order repaired and checked by reading, not compiled, not run on hardware
//
#include <hip/hip_runtime.h>
#include <hip/hip_bf16.h>

// ---------------------------------------------------------------------------
// GraphNetwork block for MI455X (gfx1250), wave32 + v_wmma_f32_16x16x32_bf16.
// D=64, H=256 fixed by the reference; N/E/G derived from in_sizes.
// ---------------------------------------------------------------------------

typedef __attribute__((ext_vector_type(16))) __bf16 v16bf;
typedef __attribute__((ext_vector_type(8)))  __bf16 v8bf;
typedef __attribute__((ext_vector_type(8)))  float  v8f;

#define DD   64
#define HH   256
#define LDH_ 264   // hidden LDS stride (bf16 elems), 528B rows (16B aligned)
#define LDO_ 68    // out LDS stride (f32 elems), 272B rows (16B aligned)

// ---- WMMA tile loaders ----------------------------------------------------
// A-tile 16x32 bf16 from LDS, row-major, per ISA layout:
// lanes 0-15: M=lane, K = [0..7]+[16..23]; lanes 16-31: M=lane-16, K=[8..15]+[24..31]
__device__ __forceinline__ v16bf ldsA16x32(const __bf16* base, int stride, int lane) {
    int row = lane & 15;
    int kb  = (lane >> 4) << 3;                  // 0 or 8
    const __bf16* p = base + row * stride + kb;
    v8bf lo = *(const v8bf*)(p);                 // K kb..kb+7
    v8bf hi = *(const v8bf*)(p + 16);            // K kb+16..kb+23
    v16bf r;
#pragma unroll
    for (int i = 0; i < 8; ++i) { r[i] = lo[i]; r[i + 8] = hi[i]; }
    return r;
}

// B-tile 32x16 bf16 from global weights stored transposed [Nout][K]:
// lane holds column n = lane&15; lanes 0-15: K=[0..7]+[16..23], lanes 16-31: K=[8..15]+[24..31]
__device__ __forceinline__ v16bf glbB32x16(const __bf16* base, int kdim, int lane) {
    int col = lane & 15;
    int kb  = (lane >> 4) << 3;                  // 0 or 8
    const __bf16* p = base + col * kdim + kb;
    v8bf lo = *(const v8bf*)(p);
    v8bf hi = *(const v8bf*)(p + 16);
    v16bf r;
#pragma unroll
    for (int i = 0; i < 8; ++i) { r[i] = lo[i]; r[i + 8] = hi[i]; }
    return r;
}

// ---- 64xK @ KxH GEMM + bias + relu -> bf16 hidden in LDS -------------------
__device__ __forceinline__ void gemm1_relu(const __bf16* A, int lda, int ksteps,
                                           const __bf16* W1t, const float* b1,
                                           __bf16* Hh) {
    int lane = threadIdx.x & 31;
    int wv   = threadIdx.x >> 5;
    int K    = ksteps << 5;
    for (int t = wv; t < 64; t += 8) {           // 4 mt x 16 nt tiles
        int mt = t >> 4, nt = t & 15;
        v8f acc = {};
        for (int kk = 0; kk < ksteps; ++kk) {
            v16bf a = ldsA16x32(A + mt * 16 * lda + kk * 32, lda, lane);
            v16bf b = glbB32x16(W1t + nt * 16 * K + kk * 32, K, lane);
            acc = __builtin_amdgcn_wmma_f32_16x16x32_bf16(
                false, a, false, b, (short)0, acc, false, false);
        }
        int col = nt * 16 + (lane & 15);
        float bv = b1[col];
        int rb = mt * 16 + ((lane >> 4) << 3);   // C layout: M = v + 8*(lane>=16)
#pragma unroll
        for (int v = 0; v < 8; ++v) {
            float x = acc[v] + bv;
            x = x > 0.f ? x : 0.f;
            Hh[(rb + v) * LDH_ + col] = (__bf16)x;
        }
    }
}

// ---- 64x256 @ 256x64 GEMM + bias -> f32 out in LDS -------------------------
__device__ __forceinline__ void gemm2_bias(const __bf16* Hh,
                                           const __bf16* W2t, const float* b2,
                                           float* O) {
    int lane = threadIdx.x & 31;
    int wv   = threadIdx.x >> 5;
    for (int t = wv; t < 16; t += 8) {           // 4 mt x 4 nt tiles
        int mt = t >> 2, nt = t & 3;
        v8f acc = {};
        for (int kk = 0; kk < 8; ++kk) {
            v16bf a = ldsA16x32(Hh + mt * 16 * LDH_ + kk * 32, LDH_, lane);
            v16bf b = glbB32x16(W2t + nt * 16 * 256 + kk * 32, 256, lane);
            acc = __builtin_amdgcn_wmma_f32_16x16x32_bf16(
                false, a, false, b, (short)0, acc, false, false);
        }
        int col = nt * 16 + (lane & 15);
        float bv = b2[col];
        int rb = mt * 16 + ((lane >> 4) << 3);
#pragma unroll
        for (int v = 0; v < 8; ++v)
            O[(rb + v) * LDO_ + col] = acc[v] + bv;
    }
}

// ---- helper: load 64 f32 row -> bf16 LDS row (8x 16B stores) ---------------
__device__ __forceinline__ void row_f32_to_bf16(const float* srow, __bf16* arow) {
#pragma unroll
    for (int i = 0; i < 8; ++i) {
        v8bf pk;
        if (srow) {
            float4 f0 = *(const float4*)(srow + i * 8);
            float4 f1 = *(const float4*)(srow + i * 8 + 4);
            pk[0] = (__bf16)f0.x; pk[1] = (__bf16)f0.y;
            pk[2] = (__bf16)f0.z; pk[3] = (__bf16)f0.w;
            pk[4] = (__bf16)f1.x; pk[5] = (__bf16)f1.y;
            pk[6] = (__bf16)f1.z; pk[7] = (__bf16)f1.w;
        } else {
#pragma unroll
            for (int j = 0; j < 8; ++j) pk[j] = (__bf16)0.f;
        }
        *(v8bf*)(arow + i * 8) = pk;
    }
}

// ---- weight transpose-convert: dst[n*K+k] = bf16(src[k*Nw+n]) --------------
__global__ void kconv(const float* __restrict__ src, __bf16* __restrict__ dst,
                      int K, int Nw) {
    int i = blockIdx.x * 256 + threadIdx.x;
    if (i < K * Nw) {
        int n = i / K, k = i - n * K;
        dst[i] = (__bf16)src[k * Nw + n];
    }
}

__global__ void kzero(float* __restrict__ p, long long n) {
    long long i = (long long)blockIdx.x * 256 + threadIdx.x;
    if (i < n) p[i] = 0.f;
}

// ---- edge kernel: 64 edges/block ------------------------------------------
__global__ __launch_bounds__(256) void kedge(
    const float* __restrict__ nf, const float* __restrict__ ef,
    const float* __restrict__ u,
    const int* __restrict__ src, const int* __restrict__ dst,
    const int* __restrict__ egid,
    const __bf16* __restrict__ W1t, const float* __restrict__ b1,
    const __bf16* __restrict__ W2t, const float* __restrict__ b2,
    float* __restrict__ efOut, float* __restrict__ aggM,
    float* __restrict__ eAggr, int E) {
    const int LDA = 264;                          // 4*64 cols + pad
    __shared__ __align__(16) unsigned char smARaw[64 * 264 * 2];
    __shared__ __align__(16) __bf16 smH[64 * LDH_];
    __shared__ int smDst[64];
    __shared__ int smGid[64];
    __bf16* smA = (__bf16*)smARaw;
    float*  smO = (float*)smARaw;                 // reuse after GEMM1

    int tid = threadIdx.x;
    long long eBase = (long long)blockIdx.x * 64;

    // Phase 1: gather+concat [ef | nf[src] | nf[dst] | u[egid]] -> bf16 LDS
    {
        int r = tid >> 2, seg = tid & 3;
        long long e = eBase + r;
        const float* srow = nullptr;
        if (e < E) {
            if (seg == 0)       srow = ef + e * DD;
            else if (seg == 1)  srow = nf + (long long)src[e] * DD;
            else if (seg == 2) { int d = dst[e]; smDst[r] = d;
                                 srow = nf + (long long)d * DD; }
            else               { int g = egid[e]; smGid[r] = g;
                                 srow = u + (long long)g * DD; }
        } else {
            if (seg == 2) smDst[r] = 0;
            if (seg == 3) smGid[r] = 0;
        }
        row_f32_to_bf16(srow, smA + r * LDA + seg * DD);
    }
    __syncthreads();

    gemm1_relu(smA, LDA, 8, W1t, b1, smH);        // K = 256
    __syncthreads();
    gemm2_bias(smH, W2t, b2, smO);                // K = 256, out 64x64
    __syncthreads();

    // Phase out: ef_new store + agg_m scatter-add
    {
        int r = tid >> 2, cs = (tid & 3) << 4;
        long long e = eBase + r;
        if (e < E) {
            const float* orow = smO + r * LDO_ + cs;
            float* grow = efOut + e * DD + cs;
#pragma unroll
            for (int j = 0; j < 4; ++j)
                *(float4*)(grow + j * 4) = *(const float4*)(orow + j * 4);
            float* arow = aggM + (long long)smDst[r] * DD + cs;
#pragma unroll
            for (int j = 0; j < 16; ++j) atomicAdd(arow + j, orow[j]);
        }
    }
    // e_aggr: edge_gid is sorted -> run-length reduce, one atomic per run
    if (tid < DD) {
        int c = tid;
        long long rem = (long long)E - eBase;
        int rows = rem < 64 ? (int)rem : 64;
        float s = 0.f; int g = smGid[0];
        for (int r2 = 0; r2 < rows; ++r2) {
            int gg = smGid[r2];
            if (gg != g) { atomicAdd(eAggr + (long long)g * DD + c, s); s = 0.f; g = gg; }
            s += smO[r2 * LDO_ + c];
        }
        atomicAdd(eAggr + (long long)g * DD + c, s);
    }
}

// ---- node kernel: 64 nodes/block ------------------------------------------
__global__ __launch_bounds__(256) void knode(
    const float* __restrict__ nf, const float* __restrict__ u,
    const int* __restrict__ ngid, const float* __restrict__ aggM,
    const __bf16* __restrict__ W1t, const float* __restrict__ b1,
    const __bf16* __restrict__ W2t, const float* __restrict__ b2,
    float* __restrict__ nfOut, float* __restrict__ nAggr, int Nn) {
    const int LDA = 200;                          // 3*64 cols + pad
    __shared__ __align__(16) unsigned char smARaw[64 * 264 * 2];
    __shared__ __align__(16) __bf16 smH[64 * LDH_];
    __shared__ int smGid[64];
    __bf16* smA = (__bf16*)smARaw;
    float*  smO = (float*)smARaw;

    int tid = threadIdx.x;
    long long base = (long long)blockIdx.x * 64;

    {
        int r = tid >> 2, seg = tid & 3;
        long long nd = base + r;
        if (seg < 3) {
            const float* srow = nullptr;
            if (nd < Nn) {
                if (seg == 0)      srow = aggM + nd * DD;
                else if (seg == 1) srow = nf + nd * DD;
                else             { int g = ngid[nd]; smGid[r] = g;
                                   srow = u + (long long)g * DD; }
            } else if (seg == 2) smGid[r] = 0;
            row_f32_to_bf16(srow, smA + r * LDA + seg * DD);
        }
    }
    __syncthreads();

    gemm1_relu(smA, LDA, 6, W1t, b1, smH);        // K = 192
    __syncthreads();
    gemm2_bias(smH, W2t, b2, smO);
    __syncthreads();

    {
        int r = tid >> 2, cs = (tid & 3) << 4;
        long long nd = base + r;
        if (nd < Nn) {
            const float* orow = smO + r * LDO_ + cs;
            float* grow = nfOut + nd * DD + cs;
#pragma unroll
            for (int j = 0; j < 4; ++j)
                *(float4*)(grow + j * 4) = *(const float4*)(orow + j * 4);
        }
    }
    if (tid < DD) {                               // node_gid sorted
        int c = tid;
        long long rem = (long long)Nn - base;
        int rows = rem < 64 ? (int)rem : 64;
        float s = 0.f; int g = smGid[0];
        for (int r2 = 0; r2 < rows; ++r2) {
            int gg = smGid[r2];
            if (gg != g) { atomicAdd(nAggr + (long long)g * DD + c, s); s = 0.f; g = gg; }
            s += smO[r2 * LDO_ + c];
        }
        atomicAdd(nAggr + (long long)g * DD + c, s);
    }
}

// ---- global kernel: one block, G<=64 graphs -------------------------------
__global__ __launch_bounds__(256) void kglobal(
    const float* __restrict__ eAggr, const float* __restrict__ nAggr,
    const float* __restrict__ u,
    const __bf16* __restrict__ W1t, const float* __restrict__ b1,
    const __bf16* __restrict__ W2t, const float* __restrict__ b2,
    float* __restrict__ uOut, int G) {
    const int LDA = 200;
    __shared__ __align__(16) unsigned char smARaw[64 * 264 * 2];
    __shared__ __align__(16) __bf16 smH[64 * LDH_];
    __bf16* smA = (__bf16*)smARaw;
    float*  smO = (float*)smARaw;

    int tid = threadIdx.x;
    {
        int r = tid >> 2, seg = tid & 3;
        if (seg < 3) {
            const float* srow = nullptr;
            if (r < G) {
                if (seg == 0)      srow = eAggr + (long long)r * DD;
                else if (seg == 1) srow = nAggr + (long long)r * DD;
                else               srow = u + (long long)r * DD;
            }
            row_f32_to_bf16(srow, smA + r * LDA + seg * DD);
        }
    }
    __syncthreads();
    gemm1_relu(smA, LDA, 6, W1t, b1, smH);        // K = 192
    __syncthreads();
    gemm2_bias(smH, W2t, b2, smO);
    __syncthreads();
    {
        int r = tid >> 2, cs = (tid & 3) << 4;
        if (r < G) {
            const float* orow = smO + r * LDO_ + cs;
            float* grow = uOut + (long long)r * DD + cs;
#pragma unroll
            for (int j = 0; j < 4; ++j)
                *(float4*)(grow + j * 4) = *(const float4*)(orow + j * 4);
        }
    }
}

// ---------------------------------------------------------------------------
extern "C" void kernel_launch(void* const* d_in, const int* in_sizes, int n_in,
                              void* d_out, int out_size, void* d_ws, size_t ws_size,
                              hipStream_t stream) {
    const float* nf  = (const float*)d_in[0];
    const float* ef  = (const float*)d_in[1];
    const float* u   = (const float*)d_in[2];
    const int* src   = (const int*)d_in[3];
    const int* dst   = (const int*)d_in[4];
    const int* ngid  = (const int*)d_in[5];
    const int* egid  = (const int*)d_in[6];
    const float* eW1 = (const float*)d_in[7];  const float* eb1 = (const float*)d_in[8];
    const float* eW2 = (const float*)d_in[9];  const float* eb2 = (const float*)d_in[10];
    const float* nW1 = (const float*)d_in[11]; const float* nb1 = (const float*)d_in[12];
    const float* nW2 = (const float*)d_in[13]; const float* nb2 = (const float*)d_in[14];
    const float* gW1 = (const float*)d_in[15]; const float* gb1 = (const float*)d_in[16];
    const float* gW2 = (const float*)d_in[17]; const float* gb2 = (const float*)d_in[18];

    const int N = in_sizes[0] / DD;
    const int E = in_sizes[1] / DD;
    const int G = in_sizes[2] / DD;

    // workspace layout (all offsets 32B-aligned)
    unsigned char* ws = (unsigned char*)d_ws;
    __bf16* eW1t = (__bf16*)(ws + 0);        // 256*256 bf16 = 131072 B
    __bf16* eW2t = (__bf16*)(ws + 131072);   //  64*256 bf16 =  32768 B
    __bf16* nW1t = (__bf16*)(ws + 163840);   // 256*192 bf16 =  98304 B
    __bf16* nW2t = (__bf16*)(ws + 262144);   //  64*256 bf16 =  32768 B
    __bf16* gW1t = (__bf16*)(ws + 294912);   // 256*192 bf16 =  98304 B
    __bf16* gW2t = (__bf16*)(ws + 393216);   //  64*256 bf16 =  32768 B
    float* aggM  = (float*)(ws + 425984);    // N*64 f32
    float* eAggr = aggM + (size_t)N * DD;    // G*64 f32
    float* nAggr = eAggr + (size_t)G * DD;   // G*64 f32

    // 1) weights -> transposed bf16
    kconv<<<(256 * 256 + 255) / 256, 256, 0, stream>>>(eW1, eW1t, 256, 256);
    kconv<<<(256 * 64  + 255) / 256, 256, 0, stream>>>(eW2, eW2t, 256, 64);
    kconv<<<(192 * 256 + 255) / 256, 256, 0, stream>>>(nW1, nW1t, 192, 256);
    kconv<<<(256 * 64  + 255) / 256, 256, 0, stream>>>(nW2, nW2t, 256, 64);
    kconv<<<(192 * 256 + 255) / 256, 256, 0, stream>>>(gW1, gW1t, 192, 256);
    kconv<<<(256 * 64  + 255) / 256, 256, 0, stream>>>(gW2, gW2t, 256, 64);

    // 2) zero accumulators (agg_m, e_aggr, n_aggr are contiguous)
    long long zn = (long long)N * DD + 2LL * G * DD;
    kzero<<<(unsigned)((zn + 255) / 256), 256, 0, stream>>>(aggM, zn);

    float* out = (float*)d_out;
    float* nfOut = out;                                     // [N,64]
    float* efOut = out + (size_t)N * DD;                    // [E,64]
    float* uOut  = out + (size_t)N * DD + (size_t)E * DD;   // [G,64]

    // 3) edge MLP + agg_m scatter + e_aggr reduce
    kedge<<<(E + 63) / 64, 256, 0, stream>>>(nf, ef, u, src, dst, egid,
                                             eW1t, eb1, eW2t, eb2,
                                             efOut, aggM, eAggr, E);
    // 4) node MLP + n_aggr reduce
    knode<<<(N + 63) / 64, 256, 0, stream>>>(nf, u, ngid, aggM,
                                             nW1t, nb1, nW2t, nb2,
                                             nfOut, nAggr, N);
    // 5) global MLP
    kglobal<<<1, 256, 0, stream>>>(eAggr, nAggr, u, gW1t, gb1, gW2t, gb2, uOut, G);
}
